// STAR_43147241455865
// MI455X (gfx1250) — matrix operand order repaired and checked by reading, not compiled, not run on hardware
//
#include <hip/hip_runtime.h>
#include <hip/hip_bf16.h>

typedef __attribute__((ext_vector_type(16))) __bf16 v16bf;
typedef __attribute__((ext_vector_type(8)))  float  v8f;
typedef __attribute__((ext_vector_type(4)))  float  vf4;

#define DEV_INLINE __device__ __forceinline__

constexpr int B_ROWS = 65536;
constexpr int D_IN   = 256;
constexpr int D_H    = 1024;
constexpr int D_F    = 512;
constexpr int MBLK   = 64;            // batch rows per workgroup
constexpr int XS_STRIDE = D_IN + 8;   // +16B/row -> fragment reads hit all 64 banks
constexpr int HS_STRIDE = D_H + 8;

union FragBF { uint4 u[2]; v16bf v; };

// A fragment (16x32 bf16, row-major source): lane<16 holds row=lane, K: k0+0..7 / k0+16..23;
// lanes 16..31 hold row=lane-16, K shifted by +8.  (ISA 7.12.2, 16-bit A matrix)
DEV_INLINE v16bf load_a_frag(const __hip_bfloat16* base, int stride, int m0, int k0, int lane) {
  const int row  = m0 + (lane & 15);
  const int koff = (lane & 16) ? 8 : 0;
  const __hip_bfloat16* p = base + row * stride + k0 + koff;
  FragBF f;
  f.u[0] = *(const uint4*)(p);        // K = k0+koff+0..7
  f.u[1] = *(const uint4*)(p + 16);   // K = k0+koff+16..23
  return f.v;
}

// B fragment (32x16 bf16) from transposed [N][K] weights: lane n holds column n0+(n&15);
// lanes 0-15 cover K=k0+0..7 / k0+16..23, lanes 16-31 cover K shifted by +8 (matches A).
DEV_INLINE v16bf load_b_frag(const __hip_bfloat16* wT, int K, int n0, int k0, int lane) {
  const int col  = n0 + (lane & 15);
  const int koff = (lane & 16) ? 8 : 0;
  const __hip_bfloat16* p = wT + col * K + k0 + koff;
  FragBF f;
  f.u[0] = *(const uint4*)(p);
  f.u[1] = *(const uint4*)(p + 16);
  return f.v;
}

// One full MLP branch for this workgroup's 64-row tile.
// MODE 0: store always to cols [0,512)       (expert c)
// MODE 1: store if tag[row] to cols [0,512)  (expert b overwrites c)
// MODE 2: store always to cols [512,1024)    (shared branch p)
template <int MODE>
DEV_INLINE void run_branch(const __hip_bfloat16* Xs, __hip_bfloat16* Hs,
                           const __hip_bfloat16* __restrict__ W1T,
                           const float* __restrict__ b1,
                           const __hip_bfloat16* __restrict__ W2T,
                           const float* __restrict__ b2,
                           const unsigned char* __restrict__ tag,
                           float* __restrict__ out,
                           int row_base, int wave, int lane) {
  // ---- layer 1: H = relu(X @ W1 + b1); 4 n-tiles per pass, A-frag reused x4 ----
  for (int g = wave; g < D_H / 64; g += 8) {
    const int n0 = g * 64;
    v8f acc[4][4] = {};                       // [n-sub][m]
    for (int k0 = 0; k0 < D_IN; k0 += 32) {
      v16bf bf[4];
#pragma unroll
      for (int j = 0; j < 4; ++j) bf[j] = load_b_frag(W1T, D_IN, n0 + j * 16, k0, lane);
#pragma unroll
      for (int m = 0; m < 4; ++m) {
        const v16bf af = load_a_frag(Xs, XS_STRIDE, m * 16, k0, lane);
#pragma unroll
        for (int j = 0; j < 4; ++j)
          acc[j][m] = __builtin_amdgcn_wmma_f32_16x16x32_bf16(false, af, false, bf[j],
                                                              (short)0, acc[j][m], false, false);
      }
    }
#pragma unroll
    for (int j = 0; j < 4; ++j) {
      const float bias = b1[n0 + j * 16 + (lane & 15)];
      const int  colH  = n0 + j * 16 + (lane & 15);
#pragma unroll
      for (int m = 0; m < 4; ++m) {
        const int rowb = m * 16 + ((lane & 16) ? 8 : 0);
#pragma unroll
        for (int r = 0; r < 8; ++r) {
          float v = acc[j][m][r] + bias;
          v = v > 0.0f ? v : 0.0f;
          Hs[(rowb + r) * HS_STRIDE + colH] = __float2bfloat16(v);
        }
      }
    }
  }
  __syncthreads();  // H complete before layer 2 reads

  // ---- layer 2: OUT = H @ W2 + b2; each wave owns one group of 4 n-tiles ----
  {
    const int n0 = wave * 64;                 // D_F/16 = 32 tiles = 8 waves * 4
    v8f acc[4][4] = {};
    for (int k0 = 0; k0 < D_H; k0 += 32) {
      v16bf bf[4];
#pragma unroll
      for (int j = 0; j < 4; ++j) bf[j] = load_b_frag(W2T, D_H, n0 + j * 16, k0, lane);
#pragma unroll
      for (int m = 0; m < 4; ++m) {
        const v16bf af = load_a_frag(Hs, HS_STRIDE, m * 16, k0, lane);
#pragma unroll
        for (int j = 0; j < 4; ++j)
          acc[j][m] = __builtin_amdgcn_wmma_f32_16x16x32_bf16(false, af, false, bf[j],
                                                              (short)0, acc[j][m], false, false);
      }
    }
#pragma unroll
    for (int j = 0; j < 4; ++j) {
      const float bias = b2[n0 + j * 16 + (lane & 15)];
      const int  col   = (MODE == 2 ? D_F : 0) + n0 + j * 16 + (lane & 15);
#pragma unroll
      for (int m = 0; m < 4; ++m) {
        const int rowb = row_base + m * 16 + ((lane & 16) ? 8 : 0);
#pragma unroll
        for (int r = 0; r < 8; ++r) {
          const int row = rowb + r;
          const float v = acc[j][m][r] + bias;
          if (MODE != 1 || tag[row])
            __builtin_nontemporal_store(v, &out[(size_t)row * (2 * D_F) + col]);
        }
      }
    }
  }
  __syncthreads();  // Hs reads done before next branch rewrites it
}

__global__ void __launch_bounds__(256)
fused_moe_mlp(const float* __restrict__ obs,
              const __hip_bfloat16* __restrict__ W1T_b, const float* __restrict__ b1_b,
              const __hip_bfloat16* __restrict__ W2T_b, const float* __restrict__ b2_b,
              const __hip_bfloat16* __restrict__ W1T_c, const float* __restrict__ b1_c,
              const __hip_bfloat16* __restrict__ W2T_c, const float* __restrict__ b2_c,
              const __hip_bfloat16* __restrict__ W1T_p, const float* __restrict__ b1_p,
              const __hip_bfloat16* __restrict__ W2T_p, const float* __restrict__ b2_p,
              const unsigned char* __restrict__ tag,
              float* __restrict__ out) {
  extern __shared__ char smem[];
  __hip_bfloat16* Xs = (__hip_bfloat16*)smem;                 // 64 x 264 bf16
  __hip_bfloat16* Hs = Xs + MBLK * XS_STRIDE;                 // 64 x 1032 bf16

  const int tid  = threadIdx.x;
  const int wave = tid >> 5;
  const int lane = tid & 31;
  const int row_base = blockIdx.x * MBLK;

  // Fill X tile: streaming (NT) f32 read, convert to bf16 in LDS. Uniform trips, no divergence.
  for (int i = tid; i < MBLK * (D_IN / 4); i += 256) {
    const int row = i >> 6;               // 64 float4 chunks per row
    const int c   = i & 63;
    const vf4 f = __builtin_nontemporal_load(
        (const vf4*)&obs[(size_t)(row_base + row) * D_IN + c * 4]);
    __hip_bfloat16* q = &Xs[row * XS_STRIDE + c * 4];
    q[0] = __float2bfloat16(f.x);
    q[1] = __float2bfloat16(f.y);
    q[2] = __float2bfloat16(f.z);
    q[3] = __float2bfloat16(f.w);
  }
  __syncthreads();

  run_branch<0>(Xs, Hs, W1T_c, b1_c, W2T_c, b2_c, tag, out, row_base, wave, lane);
  run_branch<1>(Xs, Hs, W1T_b, b1_b, W2T_b, b2_b, tag, out, row_base, wave, lane);
  run_branch<2>(Xs, Hs, W1T_p, b1_p, W2T_p, b2_p, tag, out, row_base, wave, lane);
}

// ---- prep kernels ----
__global__ void transpose_cvt(const float* __restrict__ in, __hip_bfloat16* __restrict__ out,
                              int K, int N) {
  // in: [K][N] f32 row-major -> out: [N][K] bf16
  const int total  = K * N;
  const int stride = gridDim.x * blockDim.x;
  for (int i = blockIdx.x * blockDim.x + threadIdx.x; i < total; i += stride) {
    const int k = i / N;
    const int n = i - k * N;
    out[n * K + k] = __float2bfloat16(in[i]);
  }
}

__global__ void make_tag(const float* __restrict__ obs, unsigned char* __restrict__ tag, int nrows) {
  const int stride = gridDim.x * blockDim.x;
  for (int i = blockIdx.x * blockDim.x + threadIdx.x; i < nrows; i += stride) {
    tag[i] = (obs[(size_t)i * D_IN + 2] == 1.0f && obs[(size_t)i * D_IN + 3] == 0.0f) ? 1 : 0;
  }
}

extern "C" void kernel_launch(void* const* d_in, const int* in_sizes, int n_in,
                              void* d_out, int out_size, void* d_ws, size_t ws_size,
                              hipStream_t stream) {
  (void)in_sizes; (void)n_in; (void)out_size; (void)ws_size;
  const float* obs = (const float*)d_in[0];
  // dict order: obs, then (W1,b1,W2,b2) for b, c, p
  const float* W1[3] = {(const float*)d_in[1], (const float*)d_in[5], (const float*)d_in[9]};
  const float* b1[3] = {(const float*)d_in[2], (const float*)d_in[6], (const float*)d_in[10]};
  const float* W2[3] = {(const float*)d_in[3], (const float*)d_in[7], (const float*)d_in[11]};
  const float* b2[3] = {(const float*)d_in[4], (const float*)d_in[8], (const float*)d_in[12]};

  char* ws = (char*)d_ws;
  size_t off = 0;
  auto carve = [&](size_t bytes) -> void* {
    void* p = ws + off;
    off = (off + bytes + 255) & ~(size_t)255;
    return p;
  };
  __hip_bfloat16* W1T[3];
  __hip_bfloat16* W2T[3];
  for (int i = 0; i < 3; ++i) {
    W1T[i] = (__hip_bfloat16*)carve((size_t)D_H * D_IN * sizeof(__hip_bfloat16));
    W2T[i] = (__hip_bfloat16*)carve((size_t)D_F * D_H * sizeof(__hip_bfloat16));
  }
  unsigned char* tag = (unsigned char*)carve(B_ROWS);

  for (int i = 0; i < 3; ++i) {
    transpose_cvt<<<512, 256, 0, stream>>>(W1[i], W1T[i], D_IN, D_H);
    transpose_cvt<<<512, 256, 0, stream>>>(W2[i], W2T[i], D_H, D_F);
  }
  make_tag<<<256, 256, 0, stream>>>(obs, tag, B_ROWS);

  const size_t lds_bytes = (size_t)(MBLK * XS_STRIDE + MBLK * HS_STRIDE) * sizeof(__hip_bfloat16);
  fused_moe_mlp<<<B_ROWS / MBLK, 256, lds_bytes, stream>>>(
      obs,
      W1T[0], b1[0], W2T[0], b2[0],   // b
      W1T[1], b1[1], W2T[1], b2[1],   // c
      W1T[2], b1[2], W2T[2], b2[2],   // p
      tag, (float*)d_out);
}
